// LSTMTarget_46102178955817
// MI455X (gfx1250) — matrix operand order repaired and checked by reading, not compiled, not run on hardware
//
#include <hip/hip_runtime.h>
#include <hip/hip_bf16.h>
#include <math.h>

typedef _Float16 h16 __attribute__((ext_vector_type(16)));
typedef _Float16 h8  __attribute__((ext_vector_type(8)));
typedef _Float16 h4  __attribute__((ext_vector_type(4)));
typedef float    f8  __attribute__((ext_vector_type(8)));
typedef float    f4  __attribute__((ext_vector_type(4)));

#define BATCH 1024
#define TSTEPS 512
#define FIN 40
#define HID 64
#define NOUT 8
#define APAD 72   // padded f16 row stride for A matrices
#define GPAD 260  // padded f32 row stride for gates
#define HPAD 68   // padded f32 row stride for h2 buffer

__device__ __forceinline__ f8 wmma16(h16 a, h16 b, f8 c) {
    return __builtin_amdgcn_wmma_f32_16x16x32_f16(false, a, false, b, (short)0, c,
                                                  false, false);
}

// Branch-free fast activations on the recurrence critical path.
__device__ __forceinline__ float rcp_fast(float x) { return __builtin_amdgcn_rcpf(x); }
__device__ __forceinline__ float sigf(float x) {
    return rcp_fast(1.0f + __expf(-x));
}
__device__ __forceinline__ float tanh_fast(float x) {
    // 1 - 2/(e^{2x}+1): saturates to +/-1 without branches (inf -> rcp=0)
    float e = __expf(2.0f * x);
    return 1.0f - 2.0f * rcp_fast(e + 1.0f);
}

// Load a 16x32 f16 A-fragment from LDS (row-major [16][APAD]) at K-base kb.
// Lane l<16: M=l,  halves K=kb+0..7 and kb+16..23
// Lane l>=16: M=l-16, halves K=kb+8..15 and kb+24..31
__device__ __forceinline__ h16 load_a(const _Float16* A, int lane, int kb) {
    int sel = lane >> 4;
    int m   = lane & 15;
    const h8* p0 = reinterpret_cast<const h8*>(A + m * APAD + kb + 8 * sel);
    const h8* p1 = reinterpret_cast<const h8*>(A + m * APAD + kb + 16 + 8 * sel);
    h8 lo = *p0, hi = *p1;
    return __builtin_shufflevector(lo, hi, 0, 1, 2, 3, 4, 5, 6, 7,
                                   8, 9, 10, 11, 12, 13, 14, 15);
}

// Build a 32x16 f16 B-fragment from a global row-major weight [4H][rowlen]:
// B[k][n] = W[n][k].  Lane l holds N = nb + (l&15); halves K as in load_a.
__device__ __forceinline__ h16 load_b(const float* W, int rowlen, int kmax,
                                      int nb, int kb, int lane) {
    h16 b;
    int n = nb + (lane & 15), hh = lane >> 4;
#pragma unroll
    for (int j = 0; j < 16; ++j) {
        int k = kb + (j < 8 ? 8 * hh + j : 16 + 8 * hh + (j - 8));
        float v = (k < kmax) ? W[(size_t)n * rowlen + k] : 0.0f;
        b[j] = (_Float16)v;
    }
    return b;
}

__global__ __launch_bounds__(256, 1)
void lstm2_fused_kernel(const float* __restrict__ x,
                        const float* __restrict__ Wih0, const float* __restrict__ Whh0,
                        const float* __restrict__ bih0, const float* __restrict__ bhh0,
                        const float* __restrict__ Wih1, const float* __restrict__ Whh1,
                        const float* __restrict__ bih1, const float* __restrict__ bhh1,
                        const float* __restrict__ Wlin, const float* __restrict__ blin,
                        float* __restrict__ out) {
    __shared__ _Float16 xA[16 * APAD];     // layer0 input tile (f16, K padded 40->64)
    __shared__ _Float16 hA0[16 * APAD];    // h1 state (layer0 out = layer1 in)
    __shared__ _Float16 hA1[16 * APAD];    // h2 state
    __shared__ float    gates[16 * GPAD];  // [16][256] (+pad)
    __shared__ float    h2f[16 * HPAD];    // h2 in f32 for output projection
    __shared__ float    WlinS[NOUT * HPAD];
    __shared__ float    blinS[NOUT];

    const int tid  = threadIdx.x;
    const int lane = tid & 31;
    const int w    = tid >> 5;       // wave id 0..7
    const int tb   = blockIdx.x;     // batch tile 0..63

    // ---- Preload weight B-fragments into VGPRs (time-invariant) ----
    // wf[layer][src(0=input,1=recurrent)][ntile][kchunk]
    h16 wf[2][2][2][2];
    {
        const float* Ws[2][2] = {{Wih0, Whh0}, {Wih1, Whh1}};
        const int    rl[2][2] = {{FIN, HID}, {HID, HID}};
#pragma unroll
        for (int l = 0; l < 2; ++l)
#pragma unroll
            for (int s = 0; s < 2; ++s)
#pragma unroll
                for (int nt = 0; nt < 2; ++nt)
#pragma unroll
                    for (int ch = 0; ch < 2; ++ch)
                        wf[l][s][nt][ch] = load_b(Ws[l][s], rl[l][s], rl[l][s],
                                                  32 * w + 16 * nt, 32 * ch, lane);
    }
    // Combined-bias value for this lane's gate column, per layer / N-tile.
    float biasv[2][2];
#pragma unroll
    for (int nt = 0; nt < 2; ++nt) {
        int n = 32 * w + 16 * nt + (lane & 15);
        biasv[0][nt] = bih0[n] + bhh0[n];
        biasv[1][nt] = bih1[n] + bhh1[n];
    }

    // ---- Zero-init LDS state, load W_lin / b_lin ----
    for (int i = tid; i < 16 * APAD; i += 256) {
        xA[i] = (_Float16)0.0f;
        hA0[i] = (_Float16)0.0f;
        hA1[i] = (_Float16)0.0f;
    }
    for (int i = tid; i < NOUT * HID; i += 256)
        WlinS[(i / HID) * HPAD + (i % HID)] = Wlin[i];
    if (tid < NOUT) blinS[tid] = blin[tid];

    // Cell state registers: thread t owns row (t&15), cols 4*(t>>4)..+3
    float c0r[4] = {0.f, 0.f, 0.f, 0.f};
    float c1r[4] = {0.f, 0.f, 0.f, 0.f};
    const int erow = tid & 15;
    const int ecb  = (tid >> 4) * 4;

    __syncthreads();

#pragma unroll 1
    for (int t = 0; t < TSTEPS; ++t) {
        // ---- Stage x_t tile [16][40] -> f16 LDS (cols 40..63 stay zero) ----
        if (tid < 160) {
            int row = tid / 10, j = tid % 10;
            const float* xp =
                x + ((size_t)(tb * 16 + row) * TSTEPS + t) * FIN + j * 4;
            f4 v = *reinterpret_cast<const f4*>(xp);
            h4 hv;
            hv[0] = (_Float16)v[0]; hv[1] = (_Float16)v[1];
            hv[2] = (_Float16)v[2]; hv[3] = (_Float16)v[3];
            *reinterpret_cast<h4*>(xA + row * APAD + j * 4) = hv;
            if (t + 1 < TSTEPS) __builtin_prefetch(xp + FIN, 0, 3);
        }
        __syncthreads();

        // ---- Layer 0: gates = x@Wih0^T + h1@Whh0^T + b ----
        {
            h16 ax0 = load_a(xA, lane, 0), ax1 = load_a(xA, lane, 32);
            h16 ah0 = load_a(hA0, lane, 0), ah1 = load_a(hA0, lane, 32);
#pragma unroll
            for (int nt = 0; nt < 2; ++nt) {
                float bv = biasv[0][nt];
                f8 acc = {bv, bv, bv, bv, bv, bv, bv, bv};
                acc = wmma16(ax0, wf[0][0][nt][0], acc);
                acc = wmma16(ax1, wf[0][0][nt][1], acc);
                acc = wmma16(ah0, wf[0][1][nt][0], acc);
                acc = wmma16(ah1, wf[0][1][nt][1], acc);
                int nb = 32 * w + 16 * nt;
                int lo = lane & 15, hi = lane >> 4;
#pragma unroll
                for (int r = 0; r < 8; ++r)
                    gates[(r + 8 * hi) * GPAD + nb + lo] = acc[r];
            }
        }
        __syncthreads();

        // ---- Layer 0 elementwise -> h1 (f16, into hA0) ----
        {
            const float* g = gates + erow * GPAD + ecb;
            f4 gi = *reinterpret_cast<const f4*>(g);
            f4 gf = *reinterpret_cast<const f4*>(g + 64);
            f4 gc = *reinterpret_cast<const f4*>(g + 128);
            f4 go = *reinterpret_cast<const f4*>(g + 192);
            h4 hv;
#pragma unroll
            for (int q = 0; q < 4; ++q) {
                float ig = sigf(gi[q]);
                float fg = sigf(gf[q]);
                float gg = tanh_fast(gc[q]);
                float og = sigf(go[q]);
                c0r[q] = fg * c0r[q] + ig * gg;
                hv[q] = (_Float16)(og * tanh_fast(c0r[q]));
            }
            *reinterpret_cast<h4*>(hA0 + erow * APAD + ecb) = hv;
        }
        __syncthreads();

        // ---- Layer 1: gates = h1@Wih1^T + h2@Whh1^T + b ----
        {
            h16 a10 = load_a(hA0, lane, 0), a11 = load_a(hA0, lane, 32);
            h16 a20 = load_a(hA1, lane, 0), a21 = load_a(hA1, lane, 32);
#pragma unroll
            for (int nt = 0; nt < 2; ++nt) {
                float bv = biasv[1][nt];
                f8 acc = {bv, bv, bv, bv, bv, bv, bv, bv};
                acc = wmma16(a10, wf[1][0][nt][0], acc);
                acc = wmma16(a11, wf[1][0][nt][1], acc);
                acc = wmma16(a20, wf[1][1][nt][0], acc);
                acc = wmma16(a21, wf[1][1][nt][1], acc);
                int nb = 32 * w + 16 * nt;
                int lo = lane & 15, hi = lane >> 4;
#pragma unroll
                for (int r = 0; r < 8; ++r)
                    gates[(r + 8 * hi) * GPAD + nb + lo] = acc[r];
            }
        }
        __syncthreads();

        // ---- Layer 1 elementwise -> h2 (f16 state + f32 for projection) ----
        {
            const float* g = gates + erow * GPAD + ecb;
            f4 gi = *reinterpret_cast<const f4*>(g);
            f4 gf = *reinterpret_cast<const f4*>(g + 64);
            f4 gc = *reinterpret_cast<const f4*>(g + 128);
            f4 go = *reinterpret_cast<const f4*>(g + 192);
            h4 hv;
            f4 hf;
#pragma unroll
            for (int q = 0; q < 4; ++q) {
                float ig = sigf(gi[q]);
                float fg = sigf(gf[q]);
                float gg = tanh_fast(gc[q]);
                float og = sigf(go[q]);
                c1r[q] = fg * c1r[q] + ig * gg;
                float h = og * tanh_fast(c1r[q]);
                hv[q] = (_Float16)h;
                hf[q] = h;
            }
            *reinterpret_cast<h4*>(hA1 + erow * APAD + ecb) = hv;
            *reinterpret_cast<f4*>(h2f + erow * HPAD + ecb) = hf;
        }
        __syncthreads();

        // ---- Output projection + SELU: out[b,t,:8] ----
        // (No trailing barrier needed: next conflicting write to h2f is at
        //  t+1 elementwise-L1, separated by >=3 barriers; next iteration's
        //  x-staging only touches xA.)
        if (tid < 128) {
            int row = tid >> 3, o = tid & 7;
            float s = blinS[o];
            const float* hr = h2f + row * HPAD;
            const float* wr = WlinS + o * HPAD;
#pragma unroll
            for (int k = 0; k < HID; ++k) s += hr[k] * wr[k];
            const float alpha = 1.6732632423543772f;
            const float scale = 1.0507009873554805f;
            float r = (s > 0.0f) ? scale * s : scale * alpha * (__expf(s) - 1.0f);
            out[((size_t)(tb * 16 + row) * TSTEPS + t) * NOUT + o] = r;
        }
    }
}

extern "C" void kernel_launch(void* const* d_in, const int* in_sizes, int n_in,
                              void* d_out, int out_size, void* d_ws, size_t ws_size,
                              hipStream_t stream) {
    (void)in_sizes; (void)n_in; (void)out_size; (void)d_ws; (void)ws_size;
    const float* x    = (const float*)d_in[0];
    const float* Wih0 = (const float*)d_in[1];
    const float* Whh0 = (const float*)d_in[2];
    const float* bih0 = (const float*)d_in[3];
    const float* bhh0 = (const float*)d_in[4];
    const float* Wih1 = (const float*)d_in[5];
    const float* Whh1 = (const float*)d_in[6];
    const float* bih1 = (const float*)d_in[7];
    const float* bhh1 = (const float*)d_in[8];
    const float* Wlin = (const float*)d_in[9];
    const float* blin = (const float*)d_in[10];
    float* out = (float*)d_out;

    lstm2_fused_kernel<<<BATCH / 16, 256, 0, stream>>>(
        x, Wih0, Whh0, bih0, bhh0, Wih1, Whh1, bih1, bhh1, Wlin, blin, out);
}
